// QuantumMaterialsOptimizer_71854802862782
// MI455X (gfx1250) — matrix operand order repaired and checked by reading, not compiled, not run on hardware
//
#include <hip/hip_runtime.h>
#include <hip/hip_bf16.h>
#include <math.h>

// ---------------------------------------------------------------------------
// QuantumMaterialsOptimizer for MI455X (gfx1250, wave32, WMMA + async LDS)
//
// Math simplification (exact): the annealing scan multiplies each state row by
// a per-row scalar and renormalizes -> state stays uniform 1/sqrt(QDIM). So
// probs == 1/QDIM; h_prob / coupling / transverse_field are dead and
// expected == colmean(measure_proj). Remaining work:
// params = tanh(0.7*expected + 0.3*enc), then 3 SGD steps through the
// constraint + objective nets. Big GEMMs: v_wmma_f32_16x16x32_f16 with the
// shared B panel double-buffered into LDS via global_load_async_to_lds_b128
// (one copy per workgroup serves 8 waves; DMA overlaps WMMA).
// ---------------------------------------------------------------------------

typedef __attribute__((ext_vector_type(16))) _Float16 v16h;
typedef __attribute__((ext_vector_type(8)))  float    v8f;

#define BROWS 2048
#define PDIM  4096
#define TPDIM 4086
#define CDIM  10
#define QD    1024
#define NH1   128
#define NHC   64
#define NH2   64

#define MODE_BIAS_RELU 0
#define MODE_BMAT_RELU 1
#define MODE_MASKMUL   2
#define MODE_SGD       3

// A-fragment (16x32 f16, ISA 7.12.2): lane<16 row=lane, e<8 -> K=e, e>=8 -> K=e+8;
// lanes 16-31 add +8 to K.
__device__ __forceinline__ int afrag_k(int lane, int e) {
  return (e & 7) + ((e & 8) << 1) + ((lane & 16) >> 1);
}
// B-fragment (32x16 f16): lanes 0-15 col=lane, e<8 -> K=e, e>=8 -> K=e+8;
// lanes 16-31 add +8 to K (same K order as the A-fragment).
__device__ __forceinline__ int bfrag_k(int lane, int e) {
  return (e & 7) + ((e & 8) << 1) + ((lane & 16) >> 1);
}

// -------------------- packing kernels --------------------

// Pack row-major f32 A (optionally a horizontal concat src1[K1] | src2[K2])
// into WMMA A-fragment order: [rowtile][ktile][lane][16 halves].
__global__ void pack_a_kernel(const float* __restrict__ src1, int K1,
                              const float* __restrict__ src2, int K2,
                              _Float16* __restrict__ dst, int rowtiles, int nkt) {
  int idx = blockIdx.x * blockDim.x + threadIdx.x;
  int total = rowtiles * nkt * 512;
  if (idx >= total) return;
  int e    = idx & 15;
  int lane = (idx >> 4) & 31;
  int t    = idx >> 9;
  int kt   = t % nkt;
  int rt   = t / nkt;
  int m = rt * 16 + (lane & 15);
  int k = kt * 32 + afrag_k(lane, e);
  float v = 0.0f;
  if (k < K1)                         v = src1[(size_t)m * K1 + k];
  else if (src2 && (k - K1) < K2)     v = src2[(size_t)m * K2 + (k - K1)];
  dst[idx] = (_Float16)v;
}

// Pack weight matrix into WMMA B-fragment order: [ktile][ntile][lane][16 halves].
// transpose==0: src is [K x N] row-major, leading dim ld  (val = src[k*ld+n])
// transpose==1: src is [N x K] row-major, leading dim ld  (val = src[n*ld+k])
__global__ void pack_b_kernel(const float* __restrict__ src, int ld, int transpose,
                              _Float16* __restrict__ dst, int nkt, int nnt) {
  int idx = blockIdx.x * blockDim.x + threadIdx.x;
  int total = nkt * nnt * 512;
  if (idx >= total) return;
  int e    = idx & 15;
  int lane = (idx >> 4) & 31;
  int t    = idx >> 9;
  int nt   = t % nnt;
  int kt   = t / nnt;
  int n = nt * 16 + (lane & 15);
  int k = kt * 32 + bfrag_k(lane, e);
  float v = transpose ? src[(size_t)n * ld + k] : src[(size_t)k * ld + n];
  dst[idx] = (_Float16)v;
}

// Fused backward weight: W2[k][n], k<128 -> ow1[:P,:]^T (= ow1[n*128+k]),
// k in [128,192) -> cw1^T (= cw1[n*64+(k-128)]).   K=192, N=4096.
__global__ void pack_w2_kernel(const float* __restrict__ ow1,
                               const float* __restrict__ cw1,
                               _Float16* __restrict__ dst, int nkt, int nnt) {
  int idx = blockIdx.x * blockDim.x + threadIdx.x;
  int total = nkt * nnt * 512;
  if (idx >= total) return;
  int e    = idx & 15;
  int lane = (idx >> 4) & 31;
  int t    = idx >> 9;
  int nt   = t % nnt;
  int kt   = t / nnt;
  int n = nt * 16 + (lane & 15);
  int k = kt * 32 + bfrag_k(lane, e);
  float v = (k < NH1) ? ow1[(size_t)n * NH1 + k]
                      : cw1[(size_t)n * NHC + (k - NH1)];
  dst[idx] = (_Float16)v;
}

// ---------- WMMA GEMM with async-LDS B staging ----------
// Block = 8 waves = 8 row tiles x one group of 4 n-tiles (16x64 strip per wave).
// B fragments for 2 k-tiles (8 KB) are double-buffered in LDS via
// global_load_async_to_lds_b128; staging is spread over the 8 waves (2 async
// b128 each), every wave waits only on its own ASYNCcnt, then block barrier.
// Requires: rowtiles % 8 == 0, nnt % 4 == 0, nkt % 2 == 0.
__global__ void wmma_gemm4_kernel(const _Float16* __restrict__ A,
                                  const _Float16* __restrict__ Bf,
                                  float* __restrict__ out,
                                  const float* __restrict__ biasv,
                                  const float* __restrict__ biasm,
                                  const float* __restrict__ mask,
                                  int rowtiles, int nnt, int nkt, int N,
                                  int mode, float lr) {
  __shared__ __align__(32) _Float16 sB[2][4096];   // 2 x (2 k-tiles x 4 n-tiles x 1KB)
  int w    = threadIdx.x >> 5;                     // wave 0..7
  int lane = threadIdx.x & 31;
  int ngroups = nnt >> 2;
  int rg = blockIdx.x / ngroups;                   // group of 8 row tiles
  int ng = blockIdx.x % ngroups;                   // group of 4 n-tiles
  int rt = rg * 8 + w;

  const v16h* ap = reinterpret_cast<const v16h*>(A) + (size_t)rt * nkt * 32 + lane;
  const char* bbytes = (const char*)Bf;
  unsigned lds0 = (unsigned)(uintptr_t)(&sB[0][0]);  // LDS byte offset of sB

  // Stage B fragments for k-tiles {kc*2, kc*2+1} into buffer buf.
  // 16 pieces of 512B: piece s = (kt2*8 + j*2 + h); wave w stages s==w, s==w+8.
  auto stage = [&](int kc, int buf) {
    for (int s = w; s < 16; s += 8) {
      int kt2 = s >> 3;
      int j   = (s >> 1) & 3;
      int h   = s & 1;
      unsigned lb = lds0 + (unsigned)(buf * 8192 + (kt2 * 4 + j) * 1024 + lane * 32 + h * 16);
      const char* ga = bbytes +
          ((size_t)(kc * 2 + kt2) * nnt + (size_t)(ng * 4 + j)) * 1024 + lane * 32 + h * 16;
      asm volatile("global_load_async_to_lds_b128 %0, %1, off"
                   :: "v"(lb), "v"(ga) : "memory");
    }
  };

  int nkc = nkt >> 1;
  int buf = 0;
  stage(0, 0);
  asm volatile("s_wait_asynccnt 0x0" ::: "memory");
  __syncthreads();

  v8f acc0 = {}, acc1 = {}, acc2 = {}, acc3 = {};
  for (int kc = 0; kc < nkc; ++kc) {
    if (kc + 1 < nkc) stage(kc + 1, buf ^ 1);      // overlap DMA with WMMA
    for (int kt2 = 0; kt2 < 2; ++kt2) {
      int kt = kc * 2 + kt2;
      // prefetch next A chunk into near caches (speculative; OOB dropped)
      __builtin_prefetch((const void*)(ap + (size_t)(kt + 2) * 32), 0, 3);
      v16h a = ap[(size_t)kt * 32];
      const v16h* blc = reinterpret_cast<const v16h*>(&sB[buf][kt2 * 2048]);
      v16h b0 = blc[lane];
      v16h b1 = blc[lane + 32];
      v16h b2 = blc[lane + 64];
      v16h b3 = blc[lane + 96];
      acc0 = __builtin_amdgcn_wmma_f32_16x16x32_f16(false, a, false, b0, (short)0, acc0, false, false);
      acc1 = __builtin_amdgcn_wmma_f32_16x16x32_f16(false, a, false, b1, (short)0, acc1, false, false);
      acc2 = __builtin_amdgcn_wmma_f32_16x16x32_f16(false, a, false, b2, (short)0, acc2, false, false);
      acc3 = __builtin_amdgcn_wmma_f32_16x16x32_f16(false, a, false, b3, (short)0, acc3, false, false);
    }
    asm volatile("s_wait_asynccnt 0x0" ::: "memory");  // own staging done
    __syncthreads();                                    // all waves done reading buf
    buf ^= 1;
  }

  // C/D layout: VGPR r -> row rt*16 + r + (lane>=16 ? 8:0), col = nt*16 + (lane&15)
  int mbase = rt * 16 + ((lane & 16) ? 8 : 0);
  v8f accs[4] = {acc0, acc1, acc2, acc3};
  for (int j = 0; j < 4; ++j) {
    int col = (ng * 4 + j) * 16 + (lane & 15);
    for (int r = 0; r < 8; ++r) {
      int m = mbase + r;
      size_t o = (size_t)m * N + col;
      float a = accs[j][r];
      if (mode == MODE_BIAS_RELU)      out[o] = fmaxf(a + biasv[col], 0.0f);
      else if (mode == MODE_BMAT_RELU) out[o] = fmaxf(a + biasm[o], 0.0f);
      else if (mode == MODE_MASKMUL)   out[o] = (mask[o] > 0.0f) ? a : 0.0f;
      else                             out[o] -= lr * a;   // fused SGD on params
    }
  }
}

// -------------------- small VALU kernels --------------------

// expected[p] = (1/QD) * sum_q measure_proj[q*P + p]
__global__ void colmean_kernel(const float* __restrict__ mp, float* __restrict__ exp_) {
  int p = blockIdx.x * blockDim.x + threadIdx.x;
  if (p >= PDIM) return;
  float s = 0.0f;
  for (int q = 0; q < QD; ++q) s += mp[(size_t)q * PDIM + p];
  exp_[p] = s * (1.0f / (float)QD);
}

// params = tanh(0.7*expected + 0.3*enc), enc = [target_properties | constraints]
__global__ void init_params_kernel(const float* __restrict__ tp,
                                   const float* __restrict__ cons,
                                   const float* __restrict__ exp_,
                                   float* __restrict__ params) {
  int idx = blockIdx.x * blockDim.x + threadIdx.x;
  if (idx >= BROWS * PDIM) return;
  int b = idx / PDIM, p = idx % PDIM;
  float enc = (p < TPDIM) ? tp[(size_t)b * TPDIM + p] : cons[(size_t)b * CDIM + (p - TPDIM)];
  params[idx] = tanhf(0.7f * exp_[p] + 0.3f * enc);
}

// cv = sigmoid(hC @ cw2 + cb2)     [B,10]
__global__ void cv_kernel(const float* __restrict__ hC, const float* __restrict__ cw2,
                          const float* __restrict__ cb2, float* __restrict__ cv) {
  int idx = blockIdx.x * blockDim.x + threadIdx.x;
  if (idx >= BROWS * CDIM) return;
  int b = idx / CDIM, c = idx % CDIM;
  float z = cb2[c];
  for (int j = 0; j < NHC; ++j) z += hC[(size_t)b * NHC + j] * cw2[j * CDIM + c];
  cv[idx] = 1.0f / (1.0f + expf(-z));
}

// t1[b][n] = ob1[n] + sum_c cv[b,c] * ow1[(P+c)*128 + n]  (bias matrix for h1 GEMM)
__global__ void t1_kernel(const float* __restrict__ cv, const float* __restrict__ ow1,
                          const float* __restrict__ ob1, float* __restrict__ t1) {
  int idx = blockIdx.x * blockDim.x + threadIdx.x;
  if (idx >= BROWS * NH1) return;
  int b = idx / NH1, n = idx % NH1;
  float t = ob1[n];
  for (int c = 0; c < CDIM; ++c)
    t += cv[(size_t)b * CDIM + c] * ow1[(size_t)(PDIM + c) * NH1 + n];
  t1[idx] = t;
}

// g2 = relu'(h2) * ow3
__global__ void g2_kernel(const float* __restrict__ h2, const float* __restrict__ ow3,
                          float* __restrict__ g2) {
  int idx = blockIdx.x * blockDim.x + threadIdx.x;
  if (idx >= BROWS * NH2) return;
  g2[idx] = (h2[idx] > 0.0f) ? ow3[idx % NH2] : 0.0f;
}

// gz2[b,c] = (g1[b,:] . ow1[P+c,:]) * cv*(1-cv)
__global__ void gz2_kernel(const float* __restrict__ g1, const float* __restrict__ ow1,
                           const float* __restrict__ cv, float* __restrict__ gz2) {
  int idx = blockIdx.x * blockDim.x + threadIdx.x;
  if (idx >= BROWS * CDIM) return;
  int b = idx / CDIM, c = idx % CDIM;
  float s = 0.0f;
  for (int j = 0; j < NH1; ++j)
    s += g1[(size_t)b * NH1 + j] * ow1[(size_t)(PDIM + c) * NH1 + j];
  float v = cv[idx];
  gz2[idx] = s * v * (1.0f - v);
}

// ghC[b,j] = relu'(hC) * (gz2[b,:] . cw2[j,:])
__global__ void ghc_kernel(const float* __restrict__ gz2, const float* __restrict__ cw2,
                           const float* __restrict__ hC, float* __restrict__ ghC) {
  int idx = blockIdx.x * blockDim.x + threadIdx.x;
  if (idx >= BROWS * NHC) return;
  int b = idx / NHC, j = idx % NHC;
  float r = 0.0f;
  if (hC[idx] > 0.0f) {
    for (int c = 0; c < CDIM; ++c)
      r += gz2[(size_t)b * CDIM + c] * cw2[j * CDIM + c];
  }
  ghC[idx] = r;
}

// -------------------- host launch --------------------

static inline int cdiv(int a, int b) { return (a + b - 1) / b; }

static inline void launch_gemm4(hipStream_t stream,
                                const _Float16* A, const _Float16* Bf, float* out,
                                const float* bv, const float* bm, const float* mk,
                                int rowtiles, int nnt, int nkt, int N,
                                int mode, float lr) {
  int blocks = (rowtiles / 8) * (nnt >> 2);
  wmma_gemm4_kernel<<<blocks, 256, 0, stream>>>(A, Bf, out, bv, bm, mk,
                                                rowtiles, nnt, nkt, N, mode, lr);
}

extern "C" void kernel_launch(void* const* d_in, const int* in_sizes, int n_in,
                              void* d_out, int out_size, void* d_ws, size_t ws_size,
                              hipStream_t stream) {
  const float* tp   = (const float*)d_in[0];
  const float* cons = (const float*)d_in[1];
  // d_in[2] transverse_field, d_in[3] coupling: provably unused (annealing no-op)
  const float* mp   = (const float*)d_in[4];
  const float* cw1  = (const float*)d_in[5];
  const float* cb1  = (const float*)d_in[6];
  const float* cw2  = (const float*)d_in[7];
  const float* cb2  = (const float*)d_in[8];
  const float* ow1  = (const float*)d_in[9];
  const float* ob1  = (const float*)d_in[10];
  const float* ow2  = (const float*)d_in[11];
  const float* ob2  = (const float*)d_in[12];
  const float* ow3  = (const float*)d_in[13];
  float* params = (float*)d_out;   // [B, P] -- refined in place by SGD GEMM

  char* ws = (char*)d_ws;
  size_t off = 0;
  auto alloc = [&](size_t bytes) -> void* {
    off = (off + 255) & ~(size_t)255;
    void* p = ws + off;
    off += bytes;
    return p;
  };

  const int RT = BROWS / 16;               // 128 row tiles
  float*    expected = (float*)   alloc(PDIM * 4);
  _Float16* Apar_h   = (_Float16*)alloc((size_t)BROWS * PDIM * 2);
  _Float16* cw1_h    = (_Float16*)alloc((size_t)PDIM * NHC * 2);
  _Float16* ow1a_h   = (_Float16*)alloc((size_t)PDIM * NH1 * 2);
  _Float16* ow2_h    = (_Float16*)alloc((size_t)NH1 * NH2 * 2);
  _Float16* ow2T_h   = (_Float16*)alloc((size_t)NH2 * NH1 * 2);
  _Float16* W2_h     = (_Float16*)alloc((size_t)192 * PDIM * 2);
  float*    hC       = (float*)   alloc((size_t)BROWS * NHC * 4);
  float*    cv       = (float*)   alloc((size_t)BROWS * CDIM * 4);
  float*    t1       = (float*)   alloc((size_t)BROWS * NH1 * 4);
  float*    h1       = (float*)   alloc((size_t)BROWS * NH1 * 4);
  _Float16* h1_h     = (_Float16*)alloc((size_t)BROWS * NH1 * 2);
  float*    h2       = (float*)   alloc((size_t)BROWS * NH2 * 4);
  float*    g2       = (float*)   alloc((size_t)BROWS * NH2 * 4);
  _Float16* g2_h     = (_Float16*)alloc((size_t)BROWS * NH2 * 2);
  float*    g1       = (float*)   alloc((size_t)BROWS * NH1 * 4);
  float*    gz2      = (float*)   alloc((size_t)BROWS * CDIM * 4);
  float*    ghC      = (float*)   alloc((size_t)BROWS * NHC * 4);
  _Float16* G_h      = (_Float16*)alloc((size_t)BROWS * 192 * 2);

  const int TB = 256;

  // ---- setup: expected = colmean(measure_proj); params = tanh(...) ----
  colmean_kernel<<<cdiv(PDIM, TB), TB, 0, stream>>>(mp, expected);
  init_params_kernel<<<cdiv(BROWS * PDIM, TB), TB, 0, stream>>>(tp, cons, expected, params);

  // ---- pack constant weights to WMMA B-fragments (f16) ----
  pack_b_kernel<<<cdiv(128 * 4 * 512, TB), TB, 0, stream>>>(cw1, NHC, 0, cw1_h, 128, 4);   // [4096,64]
  pack_b_kernel<<<cdiv(128 * 8 * 512, TB), TB, 0, stream>>>(ow1, NH1, 0, ow1a_h, 128, 8);  // [4096,128]
  pack_b_kernel<<<cdiv(4 * 4 * 512, TB), TB, 0, stream>>>(ow2, NH2, 0, ow2_h, 4, 4);       // [128,64]
  pack_b_kernel<<<cdiv(2 * 8 * 512, TB), TB, 0, stream>>>(ow2, NH2, 1, ow2T_h, 2, 8);      // [64,128] = ow2^T
  pack_w2_kernel<<<cdiv(6 * 256 * 512, TB), TB, 0, stream>>>(ow1, cw1, W2_h, 6, 256);      // [192,4096]

  // ---- 3 SGD refinement steps ----
  for (int it = 0; it < 3; ++it) {
    // A = params (f32 -> f16 fragments)
    pack_a_kernel<<<cdiv(RT * 128 * 512, TB), TB, 0, stream>>>(params, PDIM, nullptr, 0, Apar_h, RT, 128);

    // hC = relu(params @ cw1 + cb1)            [B,64]  K=4096
    launch_gemm4(stream, Apar_h, cw1_h, hC, cb1, nullptr, nullptr, RT, 4, 128, NHC, MODE_BIAS_RELU, 0.0f);

    cv_kernel<<<cdiv(BROWS * CDIM, TB), TB, 0, stream>>>(hC, cw2, cb2, cv);
    t1_kernel<<<cdiv(BROWS * NH1, TB), TB, 0, stream>>>(cv, ow1, ob1, t1);

    // h1 = relu(params @ ow1[:P,:] + (cv @ ow1[P:,:] + ob1))   [B,128]  K=4096
    launch_gemm4(stream, Apar_h, ow1a_h, h1, nullptr, t1, nullptr, RT, 8, 128, NH1, MODE_BMAT_RELU, 0.0f);

    // h2 = relu(h1 @ ow2 + ob2)                [B,64]  K=128
    pack_a_kernel<<<cdiv(RT * 4 * 512, TB), TB, 0, stream>>>(h1, NH1, nullptr, 0, h1_h, RT, 4);
    launch_gemm4(stream, h1_h, ow2_h, h2, ob2, nullptr, nullptr, RT, 4, 4, NH2, MODE_BIAS_RELU, 0.0f);

    // g2 = relu'(h2) * ow3;  g1 = relu'(h1) * (g2 @ ow2^T)    [B,128]  K=64
    g2_kernel<<<cdiv(BROWS * NH2, TB), TB, 0, stream>>>(h2, ow3, g2);
    pack_a_kernel<<<cdiv(RT * 2 * 512, TB), TB, 0, stream>>>(g2, NH2, nullptr, 0, g2_h, RT, 2);
    launch_gemm4(stream, g2_h, ow2T_h, g1, nullptr, nullptr, h1, RT, 8, 2, NH1, MODE_MASKMUL, 0.0f);

    // constraint-net backward (tiny)
    gz2_kernel<<<cdiv(BROWS * CDIM, TB), TB, 0, stream>>>(g1, ow1, cv, gz2);
    ghc_kernel<<<cdiv(BROWS * NHC, TB), TB, 0, stream>>>(gz2, cw2, hC, ghC);

    // params -= 0.01 * ([g1 | ghC] @ [ow1a^T ; cw1^T])        [B,4096]  K=192
    pack_a_kernel<<<cdiv(RT * 6 * 512, TB), TB, 0, stream>>>(g1, NH1, ghC, NHC, G_h, RT, 6);
    launch_gemm4(stream, G_h, W2_h, params, nullptr, nullptr, nullptr, RT, 256, 6, PDIM, MODE_SGD, 0.01f);
  }
}
